// GAT_26079041422091
// MI455X (gfx1250) — matrix operand order repaired and checked by reading, not compiled, not run on hardware
//
#include <hip/hip_runtime.h>

// ---------------- problem constants ----------------
#define NN   50000
#define EE   800000
#define ETOT 850000   // E + N self loops
#define DD   128      // feature dim = HEADS*HID = IN
#define HH   4        // heads
#define CC   32       // channels per head
#define OC   16       // classifier out
#define NEG_ATT 0.2f
#define NEG_ACT 0.01f

typedef unsigned short u16;
typedef __attribute__((ext_vector_type(16))) __bf16 v16bf;
typedef __attribute__((ext_vector_type(8)))  float  v8f;
typedef __attribute__((ext_vector_type(4)))  unsigned int u32x4;
typedef __attribute__((ext_vector_type(8)))  int i32x8;
typedef __attribute__((ext_vector_type(4)))  int i32x4;

// fp32 -> bf16 round-to-nearest-even
__device__ __forceinline__ u16 f2bf(float f) {
  unsigned u = __float_as_uint(f);
  u += 0x7fffu + ((u >> 16) & 1u);
  return (u16)(u >> 16);
}

// order-preserving float<->uint key (for atomicMax-based segment max)
__device__ __forceinline__ unsigned fkey(float f) {
  unsigned u = __float_as_uint(f);
  return (u & 0x80000000u) ? ~u : (u | 0x80000000u);
}
__device__ __forceinline__ float funkey(unsigned k) {
  unsigned u = (k & 0x80000000u) ? (k & 0x7fffffffu) : ~k;
  return __uint_as_float(u);
}

// ---------------- prep kernels ----------------
// Bt[n*K + k] = bf16(W[k*Ncols + n])   (transposed bf16 weights, K = inner dim)
__global__ void k_transpose_bf16(const float* __restrict__ W, u16* __restrict__ Bt,
                                 int K, int Ncols) {
  int i = blockIdx.x * blockDim.x + threadIdx.x;
  if (i >= K * Ncols) return;
  int n = i / K, k = i - n * K;
  Bt[n * K + k] = f2bf(W[k * Ncols + n]);
}

__global__ void k_f32_to_bf16(const float* __restrict__ X, u16* __restrict__ Y, int count) {
  int i = blockIdx.x * blockDim.x + threadIdx.x;
  if (i < count) Y[i] = f2bf(X[i]);
}

// ---------------- dual GEMM: XL = A@Wl, XR = A@Wr  (M=50000, K=N=128) -------------
// blockIdx.x = m-tile (16 rows). 8 waves, wave w owns 16-column tile w.
// A tile staged into LDS via the Tensor Data Mover (TDM); fragments reused for both
// weight matrices.
__global__ __launch_bounds__(256) void k_gemm_dual(
    const u16* __restrict__ Abf, const u16* __restrict__ BtL, const u16* __restrict__ BtR,
    float* __restrict__ XL, float* __restrict__ XR) {
  __shared__ __align__(16) u16 ldsA[16 * DD];
  const int mtile = blockIdx.x;
  const int t = threadIdx.x;

#if __has_builtin(__builtin_amdgcn_tensor_load_to_lds)
  // ---- TDM: DMA 16x128 bf16 tile (4KB) global -> LDS -------------------------
  if (t == 0) {
    unsigned long long ga = (unsigned long long)(Abf + (size_t)mtile * 16 * DD);
    unsigned ldsOff = (unsigned)(uintptr_t)(&ldsA[0]);   // addr[31:0] = LDS offset
    // D# group 0: count=1 | lds_addr | global_addr[56:0] | type=2
    u32x4 g0;
    g0.x = 1u;                                            // count=1, user descriptor
    g0.y = ldsOff;                                        // lds_addr (bytes)
    g0.z = (unsigned)(ga & 0xffffffffu);                  // global_addr[31:0]
    g0.w = (unsigned)((ga >> 32) & 0x01ffffffu) | 0x80000000u; // addr[56:32] | type=2
    // D# group 1: data_size=2B; tensor 128 x rows; tile 128 x 16; stride0=128
    int rows = NN - mtile * 16;                           // >= 16 (50000 % 16 == 0)
    i32x8 g1;
    g1.s0 = (int)(1u << 16);                              // workgroup_mask=0, data_size=1 (2B)
    g1.s1 = (int)((DD & 0xffff) << 16);                   // tensor_dim0[15:0] @ bits 63:48
    g1.s2 = (int)(((DD >> 16) & 0xffff) | ((rows & 0xffff) << 16)); // dim0 hi | dim1 lo
    g1.s3 = (int)(((rows >> 16) & 0xffff) | (DD << 16));  // dim1 hi | tile_dim0=128
    g1.s4 = 16;                                           // tile_dim1=16, tile_dim2=0
    g1.s5 = DD;                                           // tensor_dim0_stride=128
    g1.s6 = 0;
    g1.s7 = 0;
    i32x4 z4 = {0, 0, 0, 0};                              // groups 2/3 unused (2-D tensor)
    i32x8 z8 = {0, 0, 0, 0, 0, 0, 0, 0};                  // extra group slot (6-arg form)
    __builtin_amdgcn_tensor_load_to_lds(g0, g1, z4, z4, z8, 0);
    __builtin_amdgcn_s_wait_tensorcnt(0);                 // TENSORcnt is per-wave
  }
#else
  { // fallback: manual 16B-per-thread copy
    int row = t >> 4, c0 = (t & 15) << 3;
    *(uint4*)(&ldsA[row * DD + c0]) =
        *(const uint4*)(Abf + (size_t)(mtile * 16 + row) * DD + c0);
  }
#endif
  __syncthreads();

  const int wave = t >> 5, lane = t & 31;
  const int m = lane & 15;
  const int offA  = (lane < 16) ? 0 : 16;   // A: K half per half-wave
  const int koffB = (lane < 16) ? 0 : 16;   // B: K half per half-wave
  const int ncol = wave * 16 + (lane & 15);
  v8f cl = {}; v8f cr = {};
#pragma unroll
  for (int kb = 0; kb < DD; kb += 32) {
    union { uint4 q[2]; v16bf v; } a, bl, br;
    a.q[0] = *(const uint4*)(&ldsA[m * DD + kb + offA]);
    a.q[1] = *(const uint4*)(&ldsA[m * DD + kb + 8 + offA]);
    const uint4* pl = (const uint4*)(BtL + (size_t)ncol * DD + kb + koffB);
    bl.q[0] = pl[0]; bl.q[1] = pl[1];
    const uint4* pr = (const uint4*)(BtR + (size_t)ncol * DD + kb + koffB);
    br.q[0] = pr[0]; br.q[1] = pr[1];
    cl = __builtin_amdgcn_wmma_f32_16x16x32_bf16(false, a.v, false, bl.v, (short)0, cl, false, false);
    cr = __builtin_amdgcn_wmma_f32_16x16x32_bf16(false, a.v, false, br.v, (short)0, cr, false, false);
  }
  const int rbase = mtile * 16 + ((lane < 16) ? 0 : 8);
#pragma unroll
  for (int r = 0; r < 8; ++r) {
    XL[(size_t)(rbase + r) * DD + ncol] = cl[r];
    XR[(size_t)(rbase + r) * DD + ncol] = cr[r];
  }
}

// ---------------- classifier GEMM: logits = A@Wc + bc  (N=16) ----------------
__global__ __launch_bounds__(256) void k_gemm_cls(
    const u16* __restrict__ Abf, const u16* __restrict__ BtC,
    const float* __restrict__ bc, float* __restrict__ logits) {
  int wtile = (blockIdx.x * blockDim.x + threadIdx.x) >> 5;  // m-tile per wave
  int lane = threadIdx.x & 31;
  if (wtile >= NN / 16) return;
  const int m = lane & 15;
  const int offA  = (lane < 16) ? 0 : 16;
  const int koffB = (lane < 16) ? 0 : 16;
  const int ncol = lane & 15;
  v8f c = {};
#pragma unroll
  for (int kb = 0; kb < DD; kb += 32) {
    union { uint4 q[2]; v16bf v; } a, b;
    const u16* arow = Abf + (size_t)(wtile * 16 + m) * DD;
    a.q[0] = *(const uint4*)(arow + kb + offA);
    a.q[1] = *(const uint4*)(arow + kb + 8 + offA);
    const uint4* pb = (const uint4*)(BtC + (size_t)ncol * DD + kb + koffB);
    b.q[0] = pb[0]; b.q[1] = pb[1];
    c = __builtin_amdgcn_wmma_f32_16x16x32_bf16(false, a.v, false, b.v, (short)0, c, false, false);
  }
  float bias = bc[ncol];
  const int rbase = wtile * 16 + ((lane < 16) ? 0 : 8);
#pragma unroll
  for (int r = 0; r < 8; ++r)
    logits[(size_t)(rbase + r) * OC + ncol] = c[r] + bias;
}

// ---------------- per-node init: agg = bias, reset max-keys & denominators --------
__global__ void k_init_node(float* __restrict__ agg, const float* __restrict__ bias,
                            unsigned* __restrict__ emaxk, float* __restrict__ den) {
  int i = blockIdx.x * blockDim.x + threadIdx.x;
  if (i >= NN * DD) return;
  int ch = i & (DD - 1);
  agg[i] = bias[ch];
  if (ch < HH) { int n = i >> 7; emaxk[n * HH + ch] = 0u; den[n * HH + ch] = 0.f; }
}

// ---------------- edge scores: e[eid][h] = sum_c lrelu(xl[s]+xr[d],0.2)*att; segment max
__global__ void k_edge_scores(const int* __restrict__ ei, const float* __restrict__ XL,
                              const float* __restrict__ XR, const float* __restrict__ att,
                              float* __restrict__ esc, unsigned* __restrict__ emaxk) {
  int gid = blockIdx.x * blockDim.x + threadIdx.x;
  int eid = gid >> 5, lane = gid & 31;
  if (eid >= ETOT) return;
  int s = (eid < EE) ? ei[eid] : (eid - EE);
  int d = (eid < EE) ? ei[EE + eid] : (eid - EE);
  const float* xls = XL + (size_t)s * DD;
  const float* xrd = XR + (size_t)d * DD;
#pragma unroll
  for (int h = 0; h < HH; ++h) {
    float mv = xls[h * CC + lane] + xrd[h * CC + lane];
    float lr = mv > 0.f ? mv : NEG_ATT * mv;
    float p = lr * att[h * CC + lane];
    for (int o = 16; o; o >>= 1) p += __shfl_xor(p, o, 32);   // wave32 butterfly
    if (lane == 0) {
      esc[(size_t)eid * HH + h] = p;
      atomicMax(&emaxk[d * HH + h], fkey(p));
    }
  }
}

// ---------------- exp(e - max) and segment-sum denominator ----------------
__global__ void k_edge_exp(const int* __restrict__ ei, float* __restrict__ esc,
                           const unsigned* __restrict__ emaxk, float* __restrict__ den) {
  int i = blockIdx.x * blockDim.x + threadIdx.x;
  if (i >= ETOT * HH) return;
  int eid = i >> 2, h = i & 3;
  int d = (eid < EE) ? ei[EE + eid] : (eid - EE);
  float ex = __expf(esc[i] - funkey(emaxk[d * HH + h]));
  esc[i] = ex;                        // overwrite with numerator
  atomicAdd(&den[d * HH + h], ex);
}

// ---------------- aggregate: agg[dst] += xl[src] * alpha ----------------
__global__ void k_edge_aggregate(const int* __restrict__ ei, const float* __restrict__ XL,
                                 const float* __restrict__ esc, const float* __restrict__ den,
                                 float* __restrict__ agg) {
  int gid = blockIdx.x * blockDim.x + threadIdx.x;
  int eid = gid >> 5, lane = gid & 31;
  if (eid >= ETOT) return;
  int s = (eid < EE) ? ei[eid] : (eid - EE);
  int d = (eid < EE) ? ei[EE + eid] : (eid - EE);
  const float* xls = XL + (size_t)s * DD;
  // pull source row toward the caches while we compute alphas (covers the whole 512B row)
  __builtin_prefetch(xls + lane, 0, 3);
  __builtin_prefetch(xls + 64 + lane, 0, 3);
  float a = 0.f;
  if (lane < HH) a = esc[(size_t)eid * HH + lane] / (den[d * HH + lane] + 1e-16f);
  float* od = agg + (size_t)d * DD;
#pragma unroll
  for (int h = 0; h < HH; ++h) {
    float alpha = __shfl(a, h, 32);
    int ch = h * CC + lane;
    atomicAdd(&od[ch], xls[ch] * alpha);
  }
}

// ---------------- leaky_relu + recast to bf16 (feeds next GEMM) ----------------
__global__ void k_finish(const float* __restrict__ agg, u16* __restrict__ ybf,
                         float* __restrict__ yf32, int writeF32) {
  int i = blockIdx.x * blockDim.x + threadIdx.x;
  if (i >= NN * DD) return;
  float v = agg[i];
  v = v > 0.f ? v : NEG_ACT * v;
  ybf[i] = f2bf(v);
  if (writeF32) yf32[i] = v;
}

// ---------------- host orchestration ----------------
extern "C" void kernel_launch(void* const* d_in, const int* in_sizes, int n_in,
                              void* d_out, int out_size, void* d_ws, size_t ws_size,
                              hipStream_t stream) {
  (void)in_sizes; (void)n_in; (void)out_size; (void)ws_size;
  const float* x    = (const float*)d_in[0];
  const int*   ei   = (const int*)d_in[1];
  // d_in[2] = edge_weight: unused by GATv2 reference
  const float* Wl1  = (const float*)d_in[3];
  const float* Wr1  = (const float*)d_in[4];
  const float* att1 = (const float*)d_in[5];
  const float* b1   = (const float*)d_in[6];
  const float* Wl2  = (const float*)d_in[7];
  const float* Wr2  = (const float*)d_in[8];
  const float* att2 = (const float*)d_in[9];
  const float* b2   = (const float*)d_in[10];
  const float* Wc   = (const float*)d_in[11];
  const float* bc   = (const float*)d_in[12];

  float* logits = (float*)d_out;                 // [N, 16]
  float* hout   = (float*)d_out + (size_t)NN*OC; // [N, 128]

  // workspace carve (base is 256B-aligned; all segment sizes are multiples of 256)
  char* w = (char*)d_ws;
  u16* XBF  = (u16*)w;      w += (size_t)NN * DD * 2;   // bf16 activations (x / h1 / h2)
  u16* WLT1 = (u16*)w;      w += DD * DD * 2;
  u16* WRT1 = (u16*)w;      w += DD * DD * 2;
  u16* WLT2 = (u16*)w;      w += DD * DD * 2;
  u16* WRT2 = (u16*)w;      w += DD * DD * 2;
  u16* WCT  = (u16*)w;      w += OC * DD * 2 + 24 * 1024; // pad to keep 256B alignment
  float* XL   = (float*)w;  w += (size_t)NN * DD * 4;
  float* XR   = (float*)w;  w += (size_t)NN * DD * 4;
  float* AGG  = (float*)w;  w += (size_t)NN * DD * 4;
  float* ESC  = (float*)w;  w += (size_t)ETOT * HH * 4;
  unsigned* EMAXK = (unsigned*)w; w += (size_t)NN * HH * 4;
  float* DEN  = (float*)w;  w += (size_t)NN * HH * 4;

  const int TB = 256;
  const int gNodeCh = (NN * DD) / TB;          // 25000
  const int gEdgeWv = (ETOT * 32) / TB;        // 106250
  const int gEdgeH  = (ETOT * HH + TB - 1) / TB;
  const int gGemm   = NN / 16;                 // 3125 (exact)

  // prep: bf16 transposed weights + bf16 x
  k_transpose_bf16<<<(DD*DD + TB - 1)/TB, TB, 0, stream>>>(Wl1, WLT1, DD, DD);
  k_transpose_bf16<<<(DD*DD + TB - 1)/TB, TB, 0, stream>>>(Wr1, WRT1, DD, DD);
  k_transpose_bf16<<<(DD*DD + TB - 1)/TB, TB, 0, stream>>>(Wl2, WLT2, DD, DD);
  k_transpose_bf16<<<(DD*DD + TB - 1)/TB, TB, 0, stream>>>(Wr2, WRT2, DD, DD);
  k_transpose_bf16<<<(DD*OC + TB - 1)/TB, TB, 0, stream>>>(Wc,  WCT,  DD, OC);
  k_f32_to_bf16<<<gNodeCh, TB, 0, stream>>>(x, XBF, NN * DD);

  // ---- layer 1 ----
  k_gemm_dual<<<gGemm, TB, 0, stream>>>(XBF, WLT1, WRT1, XL, XR);
  k_init_node<<<gNodeCh, TB, 0, stream>>>(AGG, b1, EMAXK, DEN);
  k_edge_scores<<<gEdgeWv, TB, 0, stream>>>(ei, XL, XR, att1, ESC, EMAXK);
  k_edge_exp<<<gEdgeH, TB, 0, stream>>>(ei, ESC, EMAXK, DEN);
  k_edge_aggregate<<<gEdgeWv, TB, 0, stream>>>(ei, XL, ESC, DEN, AGG);
  k_finish<<<gNodeCh, TB, 0, stream>>>(AGG, XBF, nullptr, 0);

  // ---- layer 2 ----
  k_gemm_dual<<<gGemm, TB, 0, stream>>>(XBF, WLT2, WRT2, XL, XR);
  k_init_node<<<gNodeCh, TB, 0, stream>>>(AGG, b2, EMAXK, DEN);
  k_edge_scores<<<gEdgeWv, TB, 0, stream>>>(ei, XL, XR, att2, ESC, EMAXK);
  k_edge_exp<<<gEdgeH, TB, 0, stream>>>(ei, ESC, EMAXK, DEN);
  k_edge_aggregate<<<gEdgeWv, TB, 0, stream>>>(ei, XL, ESC, DEN, AGG);
  k_finish<<<gNodeCh, TB, 0, stream>>>(AGG, XBF, hout, 1);   // h -> d_out + bf16 for classifier

  // ---- classifier ----
  k_gemm_cls<<<(gGemm + 7) / 8, TB, 0, stream>>>(XBF, WCT, bc, logits);
}
